// DQN_71133248357081
// MI455X (gfx1250) — matrix operand order repaired and checked by reading, not compiled, not run on hardware
//
#include <hip/hip_runtime.h>
#include <hip/hip_bf16.h>

typedef __attribute__((ext_vector_type(16))) _Float16 v16h;
typedef __attribute__((ext_vector_type(8)))  _Float16 v8h;
typedef __attribute__((ext_vector_type(8)))  float    v8f;

#define NEG_SLOPE 0.2f
#define BN_EPS    1e-5f
#define TB        256

__device__ __forceinline__ float leakyf(float x){ return x > 0.f ? x : NEG_SLOPE * x; }

// Monotonic float atomic max: signed int max for >=0, unsigned min for <0.
__device__ __forceinline__ void atomicMaxFloat(float* addr, float val){
  if (val >= 0.f) atomicMax((int*)addr, __float_as_int(val));
  else            atomicMin((unsigned int*)addr, __float_as_uint(val));
}

__device__ __forceinline__ unsigned hash_u32(unsigned x){
  x ^= x >> 16; x *= 0x7feb352du; x ^= x >> 15; x *= 0x846ca68bu; x ^= x >> 16; return x;
}

// ---------------------------------------------------------------------------
// WMMA GEMM: C[nrows, NCOL] = A[nrows, K](f16) * Bf(fragment-major f16),
// f32 accumulate.  One 16-row tile per wave; K % 32 == 0, NCOL % 16 == 0.
//
// A fragment (16-bit A 16x32, §7.12.2): lanes 0-15 row M=lane hold K 0..7 &
// 16..23; lanes 16-31 hold K 8..15 & 24..31  -> two contiguous b128 loads.
// B fragment: lane -> col = lane&15; lanes 0-15 K 0..7 & 16..23, lanes 16-31 K 8..15 & 24..31,
// pre-packed fragment-major so each lane reads 16 contiguous halves.
// C/D (f32 16x16): lane -> col, VGPR v -> row v (+8 for upper half-wave).
// ---------------------------------------------------------------------------
template<int K, int NCOL>
__global__ void gemm_wmma_f16(const _Float16* __restrict__ A,
                              const _Float16* __restrict__ Bf,
                              float* __restrict__ C, int nrows){
  constexpr int NT = NCOL / 16;
  const int wave = threadIdx.x >> 5;
  const int lane = threadIdx.x & 31;
  const int r0 = (blockIdx.x * (blockDim.x >> 5) + wave) * 16;
  if (r0 >= nrows) return;                 // wave-uniform: EXEC stays all-ones
  const int  lo = lane & 15;
  const bool hi = lane >= 16;

  v8f acc[NT];
#pragma unroll
  for (int t = 0; t < NT; ++t)
#pragma unroll
    for (int i = 0; i < 8; ++i) acc[t][i] = 0.f;

#pragma unroll
  for (int kk = 0; kk < K; kk += 32){
    v16h a;
    const _Float16* ap = A + (size_t)(r0 + lo) * K + kk + (hi ? 8 : 0);
    *(v8h*)&a       = *(const v8h*)ap;         // halves 0..7
    *((v8h*)&a + 1) = *(const v8h*)(ap + 16);  // halves 8..15
#pragma unroll
    for (int t = 0; t < NT; ++t){
      // fragment index f = (kk/32)*NT + t; per-lane 16 contiguous halves
      const _Float16* bp = Bf + (((size_t)((kk >> 5) * NT + t) * 32 + lane) << 4);
      v16h b;
      *(v8h*)&b       = *(const v8h*)bp;
      *((v8h*)&b + 1) = *(const v8h*)(bp + 8);
      acc[t] = __builtin_amdgcn_wmma_f32_16x16x32_f16(
                   false, a, false, b, (short)0, acc[t], false, false);
    }
  }
#pragma unroll
  for (int t = 0; t < NT; ++t)
#pragma unroll
    for (int v = 0; v < 8; ++v){
      int row = r0 + v + (hi ? 8 : 0);
      int col = t * 16 + lo;
      C[(size_t)row * NCOL + col] = acc[t][v];
    }
}

// ---------------------------------------------------------------------------
// Packing.
// ---------------------------------------------------------------------------
// Weights f32[Ksrc, ncol] -> f16 fragment-major, zero-padded to Kdst rows.
// Element layout: Bf[((kt*NT + tt)*32 + lane)*16 + i] = W[k, col] where
// k = kt*32 + (lane<16 ? 0:8) + (i<8 ? i : 8+i), col = tt*16 + (lane&15).
__global__ void pack_b_frag(const float* __restrict__ W, _Float16* __restrict__ Bf,
                            int Ksrc, int Kdst, int ncol){
  int idx = blockIdx.x * blockDim.x + threadIdx.x;
  if (idx >= Kdst * ncol) return;
  int i    = idx & 15;
  int lane = (idx >> 4) & 31;
  int f    = idx >> 9;
  int NT   = ncol >> 4;
  int kt   = f / NT;
  int tt   = f - kt * NT;
  int k    = kt * 32 + (lane < 16 ? 0 : 8) + (i < 8 ? i : 8 + i);
  int col  = tt * 16 + (lane & 15);
  Bf[idx] = (_Float16)(k < Ksrc ? W[k * ncol + col] : 0.f);
}

// Activations f32[n, Ksrc] -> f16[n, Kdst] row-major, zero K-padding.
__global__ void pack_act_f16(const float* __restrict__ X, _Float16* __restrict__ Xh,
                             int n, int Ksrc, int Kdst){
  int t = blockIdx.x * blockDim.x + threadIdx.x;
  if (t >= n * Kdst) return;
  int k = t % Kdst, node = t / Kdst;
  Xh[t] = (_Float16)(k < Ksrc ? X[(size_t)node * Ksrc + k] : 0.f);
}

// ---------------------------------------------------------------------------
// GAT attention pipeline (with self-loops).
// ---------------------------------------------------------------------------
__global__ void attn_coef(const float* __restrict__ h, const float* __restrict__ aS,
                          const float* __restrict__ aD, float* es, float* ed,
                          int n, int H, int C){
  int t = blockIdx.x * blockDim.x + threadIdx.x;
  if (t >= n * H) return;
  int hd = t % H;
  const float* hp = h + (size_t)(t / H) * H * C + hd * C;
  float s = 0.f, d = 0.f;
  for (int c = 0; c < C; ++c){ float v = hp[c]; s += v * aS[hd*C+c]; d += v * aD[hd*C+c]; }
  es[t] = s; ed[t] = d;
}

__global__ void att_self_max(const float* es, const float* ed, float* mx, int nH){
  int t = blockIdx.x * blockDim.x + threadIdx.x;
  if (t < nH) mx[t] = leakyf(es[t] + ed[t]);      // self-loop initializes max
}

__global__ void edge_max_k(const int* __restrict__ ei, const float* es, const float* ed,
                           float* mx, int E, int H){
  int t = blockIdx.x * blockDim.x + threadIdx.x;
  if (t >= E * H) return;
  int e = t / H, hd = t % H;
  int s = ei[e], d = ei[E + e];
  atomicMaxFloat(&mx[d * H + hd], leakyf(es[s * H + hd] + ed[d * H + hd]));
}

__global__ void att_self_sum(const float* es, const float* ed, const float* mx,
                             float* sm, int nH){
  int t = blockIdx.x * blockDim.x + threadIdx.x;
  if (t < nH) sm[t] = __expf(leakyf(es[t] + ed[t]) - mx[t]);
}

__global__ void edge_sum_k(const int* __restrict__ ei, const float* es, const float* ed,
                           const float* mx, float* sm, int E, int H){
  int t = blockIdx.x * blockDim.x + threadIdx.x;
  if (t >= E * H) return;
  int e = t / H, hd = t % H;
  int s = ei[e], dh = ei[E + e] * H + hd;
  atomicAdd(&sm[dh], __expf(leakyf(es[s * H + hd] + ed[dh]) - mx[dh]));
}

__global__ void agg_self(const float* __restrict__ h, const float* es, const float* ed,
                         const float* mx, const float* sm, const float* __restrict__ bias,
                         float* out, int n, int H, int C){
  int HC = H * C;
  int t = blockIdx.x * blockDim.x + threadIdx.x;
  if (t >= n * HC) return;
  int node = t / HC, c = t % HC, hd = c / C;
  int nh = node * H + hd;
  float alpha = __expf(leakyf(es[nh] + ed[nh]) - mx[nh]) / sm[nh];
  out[t] = bias[c] + alpha * h[t];
}

__global__ void edge_agg_k(const int* __restrict__ ei, const float* __restrict__ h,
                           const float* es, const float* ed, const float* mx,
                           const float* sm, float* out, int E, int H, int C){
  int HC = H * C;
  int t = blockIdx.x * blockDim.x + threadIdx.x;
  if (t >= E * HC) return;
  int e = t / HC, c = t % HC, hd = c / C;
  int s = ei[e], d = ei[E + e];
  int dh = d * H + hd;
  float alpha = __expf(leakyf(es[s * H + hd] + ed[dh]) - mx[dh]) / sm[dh];
  atomicAdd(&out[(size_t)d * HC + c], alpha * h[(size_t)s * HC + c]);
}

// ---------------------------------------------------------------------------
// BatchNorm (training: biased variance), ELU, dropout, f16 repack.
// stats layout: [0,HC)=sum [HC,2HC)=sumsq [2HC,3HC)=scale [3HC,4HC)=shift
// ---------------------------------------------------------------------------
__global__ void zero_f32(float* p, int n){
  int t = blockIdx.x * blockDim.x + threadIdx.x;
  if (t < n) p[t] = 0.f;
}

__global__ void bn_stats_k(const float* __restrict__ x, float* stats, int n, int HC){
  int tid = blockIdx.x * blockDim.x + threadIdx.x;
  int total = gridDim.x * blockDim.x;
  int ch = tid % HC;
  int rstride = total / HC;
  float s = 0.f, ss = 0.f;
  for (int r = tid / HC; r < n; r += rstride){
    __builtin_prefetch(&x[(size_t)(r + rstride) * HC + ch], 0, 1);
    float v = x[(size_t)r * HC + ch]; s += v; ss += v * v;
  }
  atomicAdd(&stats[ch], s);
  atomicAdd(&stats[HC + ch], ss);
}

__global__ void bn_final_k(float* stats, const float* g, const float* bt, int HC, int n){
  int t = threadIdx.x;
  if (t >= HC) return;
  float mean = stats[t] / (float)n;
  float var  = stats[HC + t] / (float)n - mean * mean;
  float sc   = g[t] * rsqrtf(var + BN_EPS);
  stats[2 * HC + t] = sc;
  stats[3 * HC + t] = bt[t] - mean * sc;
}

__global__ void bn_elu_drop_pack(const float* __restrict__ x, const float* stats,
                                 _Float16* __restrict__ out16, int n, int HC,
                                 int Kdst, unsigned seed){
  int t = blockIdx.x * blockDim.x + threadIdx.x;
  if (t >= n * Kdst) return;
  int k = t % Kdst, node = t / Kdst;
  float v = 0.f;
  if (k < HC){
    v = x[(size_t)node * HC + k] * stats[2 * HC + k] + stats[3 * HC + k];
    v = v > 0.f ? v : __expf(v) - 1.f;                         // ELU
    unsigned hsh = hash_u32((unsigned)(node * HC + k) ^ (seed * 0x9e3779b9u));
    v = (hsh & 0x8000u) ? v * 2.f : 0.f;                       // dropout p=0.5
  }
  out16[t] = (_Float16)v;
}

__global__ void bn_apply_f32(const float* __restrict__ x, const float* stats,
                             float* out, int n, int HC){
  int t = blockIdx.x * blockDim.x + threadIdx.x;
  if (t >= n * HC) return;
  int k = t % HC;
  out[t] = x[t] * stats[2 * HC + k] + stats[3 * HC + k];
}

// conv4: single output column -> plain dot product.
__global__ void dot_w4(const float* __restrict__ x, const float* __restrict__ W,
                       float* h4, int n){
  int t = blockIdx.x * blockDim.x + threadIdx.x;
  if (t >= n) return;
  const float* xp = x + (size_t)t * 32;
  float s = 0.f;
#pragma unroll
  for (int c = 0; c < 32; ++c) s += xp[c] * W[c];
  h4[t] = s;
}

// ---------------------------------------------------------------------------
// Final tanh + softmax over all N nodes.
// ---------------------------------------------------------------------------
__global__ void tanh_k(const float* a, float* out, int n){
  int t = blockIdx.x * blockDim.x + threadIdx.x;
  if (t < n) out[t] = tanhf(a[t]);
}

__global__ void red_init_k(float* red){
  if (threadIdx.x == 0){ red[0] = -1e30f; red[1] = 0.f; }
}

__global__ void reduce_max_k(const float* __restrict__ x, float* red, int n){
  __shared__ float sd[TB];
  int t = blockIdx.x * TB + threadIdx.x;
  sd[threadIdx.x] = (t < n) ? x[t] : -1e30f;
  __syncthreads();
  for (int s = TB / 2; s > 0; s >>= 1){
    if (threadIdx.x < s) sd[threadIdx.x] = fmaxf(sd[threadIdx.x], sd[threadIdx.x + s]);
    __syncthreads();
  }
  if (threadIdx.x == 0) atomicMaxFloat(&red[0], sd[0]);
}

__global__ void reduce_sumexp_k(const float* __restrict__ x, float* red, int n){
  __shared__ float sd[TB];
  int t = blockIdx.x * TB + threadIdx.x;
  sd[threadIdx.x] = (t < n) ? __expf(x[t] - red[0]) : 0.f;
  __syncthreads();
  for (int s = TB / 2; s > 0; s >>= 1){
    if (threadIdx.x < s) sd[threadIdx.x] += sd[threadIdx.x + s];
    __syncthreads();
  }
  if (threadIdx.x == 0) atomicAdd(&red[1], sd[0]);
}

__global__ void prob_k(const float* __restrict__ logits, const float* red,
                       float* prob, int n){
  int t = blockIdx.x * blockDim.x + threadIdx.x;
  if (t < n) prob[t] = __expf(logits[t] - red[0]) / red[1];
}

// ---------------------------------------------------------------------------
extern "C" void kernel_launch(void* const* d_in, const int* in_sizes, int n_in,
                              void* d_out, int out_size, void* d_ws, size_t ws_size,
                              hipStream_t stream) {
  const float* x   = (const float*)d_in[0];
  const int*   ei  = (const int*)  d_in[1];
  const float* W1  = (const float*)d_in[2];
  const float* as1 = (const float*)d_in[3];
  const float* ad1 = (const float*)d_in[4];
  const float* b1  = (const float*)d_in[5];
  const float* g1  = (const float*)d_in[6];
  const float* bt1 = (const float*)d_in[7];
  const float* W2  = (const float*)d_in[8];
  const float* as2 = (const float*)d_in[9];
  const float* ad2 = (const float*)d_in[10];
  const float* b2  = (const float*)d_in[11];
  const float* g2  = (const float*)d_in[12];
  const float* bt2 = (const float*)d_in[13];
  const float* W3  = (const float*)d_in[14];
  const float* as3 = (const float*)d_in[15];
  const float* ad3 = (const float*)d_in[16];
  const float* b3  = (const float*)d_in[17];
  const float* g3  = (const float*)d_in[18];
  const float* bt3 = (const float*)d_in[19];
  const float* W4  = (const float*)d_in[20];
  const float* as4 = (const float*)d_in[21];
  const float* ad4 = (const float*)d_in[22];
  const float* b4  = (const float*)d_in[23];
  (void)n_in; (void)out_size; (void)ws_size;

  const int N = in_sizes[0] / 4;
  const int E = in_sizes[1] / 2;

  // Scratch layout
  char* wsb = (char*)d_ws;
  size_t off = 0;
  auto take = [&](size_t bytes)->char*{
    char* p = wsb + off;
    off = (off + bytes + 255) & ~(size_t)255;
    return p;
  };
  float*    bufH  = (float*)   take((size_t)N * 64 * 4);  // pre-agg features h
  float*    bufO  = (float*)   take((size_t)N * 64 * 4);  // aggregated output
  _Float16* f16A  = (_Float16*)take((size_t)N * 64 * 2);  // f16 GEMM input
  _Float16* wh    = (_Float16*)take(64 * 64 * 2);         // f16 fragment-major W
  float*    es    = (float*)   take((size_t)N * 4 * 4);
  float*    ed    = (float*)   take((size_t)N * 4 * 4);
  float*    mx    = (float*)   take((size_t)N * 4 * 4);
  float*    sm    = (float*)   take((size_t)N * 4 * 4);
  float*    stats = (float*)   take(256 * 4);
  float*    red   = (float*)   take(2 * 4);
  float*    h4    = (float*)   take((size_t)N * 4);

  auto G = [](long long n, int b){ return (unsigned)((n + b - 1) / b); };
  const unsigned gemmGrid = G((N + 15) / 16, TB / 32);   // 8 waves/block, 1 tile/wave

  auto run_gat = [&](const float* h, const float* aS, const float* aD,
                     const float* bias, float* out, int H, int C){
    int HC = H * C;
    attn_coef   <<<G((long long)N*H,  TB), TB, 0, stream>>>(h, aS, aD, es, ed, N, H, C);
    att_self_max<<<G((long long)N*H,  TB), TB, 0, stream>>>(es, ed, mx, N*H);
    edge_max_k  <<<G((long long)E*H,  TB), TB, 0, stream>>>(ei, es, ed, mx, E, H);
    att_self_sum<<<G((long long)N*H,  TB), TB, 0, stream>>>(es, ed, mx, sm, N*H);
    edge_sum_k  <<<G((long long)E*H,  TB), TB, 0, stream>>>(ei, es, ed, mx, sm, E, H);
    agg_self    <<<G((long long)N*HC, TB), TB, 0, stream>>>(h, es, ed, mx, sm, bias, out, N, H, C);
    edge_agg_k  <<<G((long long)E*HC, TB), TB, 0, stream>>>(ei, h, es, ed, mx, sm, out, E, H, C);
  };
  auto run_bn = [&](const float* xin, const float* g, const float* bt, int HC){
    zero_f32   <<<1, 256, 0, stream>>>(stats, 2 * HC);
    bn_stats_k <<<512, TB, 0, stream>>>(xin, stats, N, HC);
    bn_final_k <<<1, 64, 0, stream>>>(stats, g, bt, HC, N);
  };

  // ---- Layer 1: in=4 (pad K to 32), heads=4, out=16 ----
  pack_b_frag <<<G(32*64, TB), TB, 0, stream>>>(W1, wh, 4, 32, 64);
  pack_act_f16<<<G((long long)N*32, TB), TB, 0, stream>>>(x, f16A, N, 4, 32);
  gemm_wmma_f16<32, 64><<<gemmGrid, TB, 0, stream>>>(f16A, wh, bufH, N);
  run_gat(bufH, as1, ad1, b1, bufO, 4, 16);
  run_bn(bufO, g1, bt1, 64);
  bn_elu_drop_pack<<<G((long long)N*64, TB), TB, 0, stream>>>(bufO, stats, f16A, N, 64, 64, 1u);

  // ---- Layer 2: in=64, heads=4, out=16 ----
  pack_b_frag<<<G(64*64, TB), TB, 0, stream>>>(W2, wh, 64, 64, 64);
  gemm_wmma_f16<64, 64><<<gemmGrid, TB, 0, stream>>>(f16A, wh, bufH, N);
  run_gat(bufH, as2, ad2, b2, bufO, 4, 16);
  run_bn(bufO, g2, bt2, 64);
  bn_elu_drop_pack<<<G((long long)N*64, TB), TB, 0, stream>>>(bufO, stats, f16A, N, 64, 64, 2u);

  // ---- Layer 3: in=64, heads=1, out=32 ----
  pack_b_frag<<<G(64*32, TB), TB, 0, stream>>>(W3, wh, 64, 64, 32);
  gemm_wmma_f16<64, 32><<<gemmGrid, TB, 0, stream>>>(f16A, wh, bufH, N);
  run_gat(bufH, as3, ad3, b3, bufO, 1, 32);
  run_bn(bufO, g3, bt3, 32);
  bn_apply_f32<<<G((long long)N*32, TB), TB, 0, stream>>>(bufO, stats, bufH, N, 32);

  // ---- Layer 4: in=32, heads=1, out=1 ----
  dot_w4<<<G(N, TB), TB, 0, stream>>>(bufH, W4, h4, N);
  run_gat(h4, as4, ad4, b4, bufO, 1, 1);

  // ---- tanh + global softmax ----
  float* logits = (float*)d_out;
  float* prob   = logits + N;
  tanh_k         <<<G(N, TB), TB, 0, stream>>>(bufO, logits, N);
  red_init_k     <<<1, 32, 0, stream>>>(red);
  reduce_max_k   <<<G(N, TB), TB, 0, stream>>>(logits, red, N);
  reduce_sumexp_k<<<G(N, TB), TB, 0, stream>>>(logits, red, N);
  prob_k         <<<G(N, TB), TB, 0, stream>>>(logits, red, prob, N);
}
